// Seq2Seq_12567074308983
// MI455X (gfx1250) — matrix operand order repaired and checked by reading, not compiled, not run on hardware
//
#include <hip/hip_runtime.h>
#include <math.h>

typedef __attribute__((ext_vector_type(16))) _Float16 v16h;
typedef __attribute__((ext_vector_type(8)))  float    v8f;

#define NB   32      // batch
#define HH   512     // hidden
#define EE   512     // embed
#define VV   10000   // vocab
#define TO   64      // To
#define TT   65      // To + 1
#define TI   512     // encoder time

// ---------------------------------------------------------------------------
// WMMA fragment helpers (layouts per CDNA5 ISA 7.12.2, f16 16x16x32)
// ---------------------------------------------------------------------------
__device__ inline v16h load_a_frag(const float* __restrict__ a_row, int kc, int hi)
{
    // lane<16 holds K = {0..7,16..23}+kc ; lane>=16 holds K = {8..15,24..31}+kc
    v16h a;
#pragma unroll
    for (int p = 0; p < 8; ++p) {
        int k = kc + ((p >> 2) << 4) + (hi << 3) + ((p & 3) << 1);
        a[2 * p]     = (_Float16)a_row[k];
        a[2 * p + 1] = (_Float16)a_row[k + 1];
    }
    return a;
}

__device__ inline v16h load_b_frag(const float* __restrict__ w_row, int kc, int hi)
{
    // lane<16 holds K = {0..7,16..23}+kc ; lane>=16 holds K = {8..15,24..31}+kc (col = lane&15)
    v16h b;
    int kb = kc + (hi << 3);
#pragma unroll
    for (int q = 0; q < 16; ++q)
        b[q] = (_Float16)w_row[kb + ((q >> 3) << 4) + (q & 7)];
    return b;
}

__device__ inline v8f wmma16(v16h a, v16h b, v8f c)
{
    return __builtin_amdgcn_wmma_f32_16x16x32_f16(
        false, a, false, b, (short)0, c, false, false);
}

// Accumulate one GEMM pass over K into a 2(M-tiles) x NACT(N-tiles) register
// block. NACT is compile-time so the K-loop carries no EXEC manipulation.
template <int NACT>
__device__ inline void accum_pass(const float* __restrict__ A, int lda, int K,
                                  const float* __restrict__ W, int ldw,
                                  int ntile0, int row, int hi,
                                  v8f (&acc)[2][2])
{
    const float* a0 = A + (size_t)(row) * lda;
    const float* a1 = A + (size_t)(16 + row) * lda;
    const float* w0 = W + (size_t)((ntile0 << 4) + row) * ldw;
    const float* w1 = w0 + (size_t)16 * ldw;
    for (int kc = 0; kc < K; kc += 32) {
        v16h fa0 = load_a_frag(a0, kc, hi);
        v16h fa1 = load_a_frag(a1, kc, hi);
        v16h fb0 = load_b_frag(w0, kc, hi);
        acc[0][0] = wmma16(fa0, fb0, acc[0][0]);
        acc[1][0] = wmma16(fa1, fb0, acc[1][0]);
        if (NACT > 1) {                       // compile-time
            v16h fb1 = load_b_frag(w1, kc, hi);
            acc[0][1] = wmma16(fa0, fb1, acc[0][1]);
            acc[1][1] = wmma16(fa1, fb1, acc[1][1]);
        }
    }
}

// Fused GEMM, M fixed at 32 (2 M-tiles per wave), 2 N-tiles per wave:
//   out[m, n] = act( A1[m,:K1]·W1[n,:K1] + A2[m,:K2]·W2[n,:K2] + bias1[n] + bias2[n] )
__global__ __launch_bounds__(256)
void gemm_wmma_kernel(const float* __restrict__ A1, int lda1, int K1,
                      const float* __restrict__ A2, int lda2, int K2,
                      const float* __restrict__ W1, int ldw1,
                      const float* __restrict__ W2, int ldw2,
                      const float* __restrict__ bias1,
                      const float* __restrict__ bias2,
                      float* __restrict__ out, long out_row_stride,
                      int Ncols, int act_tanh)
{
    int wave = (int)((blockIdx.x * blockDim.x + threadIdx.x) >> 5);
    int lane = (int)(threadIdx.x & 31);
    int tiles_n = Ncols >> 4;
    int ngroups = (tiles_n + 1) >> 1;
    if (wave >= ngroups) return;          // wave-uniform guard, EXEC stays full

    int ntile0 = wave << 1;
    int nact   = tiles_n - ntile0; if (nact > 2) nact = 2;
    int row = lane & 15;
    int hi  = lane >> 4;

    v8f acc[2][2] = {};
    if (nact == 2) {                      // wave-uniform branch, outside K-loops
        accum_pass<2>(A1, lda1, K1, W1, ldw1, ntile0, row, hi, acc);
        if (A2) accum_pass<2>(A2, lda2, K2, W2, ldw2, ntile0, row, hi, acc);
    } else {
        accum_pass<1>(A1, lda1, K1, W1, ldw1, ntile0, row, hi, acc);
        if (A2) accum_pass<1>(A2, lda2, K2, W2, ldw2, ntile0, row, hi, acc);
    }

    // C/D layout: vgpr i -> M = 16*mt + 8*hi + i ; N = lane&15 within tile
    int mbase = hi << 3;
    for (int nj = 0; nj < nact; ++nj) {   // wave-uniform bound
        int ncol = ((ntile0 + nj) << 4) + row;
        float b = 0.f;
        if (bias1) b += bias1[ncol];
        if (bias2) b += bias2[ncol];
#pragma unroll
        for (int mt = 0; mt < 2; ++mt) {
#pragma unroll
            for (int i = 0; i < 8; ++i) {
                float v = acc[mt][nj][i] + b;
                if (act_tanh) v = tanhf(v);
                out[(size_t)(mt * 16 + mbase + i) * (size_t)out_row_stride + ncol] = v;
            }
        }
    }
}

// ---------------------------------------------------------------------------
// Pointwise / glue kernels
// ---------------------------------------------------------------------------
__global__ void zero_kernel(float* __restrict__ p, int n)
{
    int i = blockIdx.x * blockDim.x + threadIdx.x;
    if (i < n) p[i] = 0.f;
}

__global__ void build_rnn_in_kernel(const int* __restrict__ tok,
                                    const float* __restrict__ emb,
                                    const float* __restrict__ att_ctx,
                                    float* __restrict__ rnn_in, int t)
{
    int idx = blockIdx.x * blockDim.x + threadIdx.x;
    if (idx >= NB * 1024) return;
    int n = idx >> 10, k = idx & 1023;
    float v;
    if (k < EE) {
        int token = (t == 0) ? 1 /*SOS*/ : tok[n * TO + (t - 1)];
        v = emb[(size_t)token * EE + k];
    } else {
        v = att_ctx[n * HH + (k - EE)];
    }
    rnn_in[idx] = v;
}

__global__ void lstm_pw_kernel(const float* __restrict__ gates,
                               float* __restrict__ h, float* __restrict__ c)
{
    int idx = blockIdx.x * blockDim.x + threadIdx.x;
    if (idx >= NB * HH) return;
    int n = idx >> 9, j = idx & 511;
    const float* g = gates + (size_t)n * 2048;
    float gi = g[j], gf = g[512 + j], gg = g[1024 + j], go = g[1536 + j];
    float si = 1.f / (1.f + expf(-gi));
    float sf = 1.f / (1.f + expf(-gf));
    float so = 1.f / (1.f + expf(-go));
    float c2 = sf * c[idx] + si * tanhf(gg);
    float h2 = so * tanhf(c2);
    c[idx] = c2;
    h[idx] = h2;
}

__global__ __launch_bounds__(512)
void attn_kernel(const float* __restrict__ enc, const float* __restrict__ r_in,
                 float* __restrict__ att_ctx, float* __restrict__ mlp_in)
{
    __shared__ float s_r[HH];
    __shared__ float s_sc[TI];
    __shared__ float s_max, s_sum;
    int n = blockIdx.x, tid = threadIdx.x;

    s_r[tid] = r_in[n * HH + tid];
    __syncthreads();

    const float* e  = enc + (size_t)n * TI * HH;
    const float* er = e + (size_t)tid * HH;
    float acc = 0.f;
    for (int k = 0; k < HH; ++k) acc += s_r[k] * er[k];
    s_sc[tid] = acc;
    __syncthreads();

    if (tid == 0) {
        float m = -1e30f;
        for (int t = 0; t < TI; ++t) m = fmaxf(m, s_sc[t]);
        s_max = m;
    }
    __syncthreads();
    float w = expf(s_sc[tid] - s_max);
    s_sc[tid] = w;
    __syncthreads();
    if (tid == 0) {
        float s = 0.f;
        for (int t = 0; t < TI; ++t) s += s_sc[t];
        s_sum = s;
    }
    __syncthreads();
    float inv = 1.f / s_sum;

    float a = 0.f;
    for (int t = 0; t < TI; ++t) a += s_sc[t] * e[(size_t)t * HH + tid];
    a *= inv;

    att_ctx[n * HH + tid]        = a;
    mlp_in[n * 1024 + 512 + tid] = a;
    mlp_in[n * 1024 + tid]       = s_r[tid];
}

__global__ __launch_bounds__(256)
void nll_kernel(const float* __restrict__ y_all, const int* __restrict__ tok,
                float* __restrict__ nll)
{
    int b = blockIdx.x;            // 0 .. NB*TT-1
    int n = b / TT, t = b % TT;
    const float* y = y_all + ((size_t)n * TT + t) * VV;
    int tid = threadIdx.x;
    __shared__ float red[256];

    float m = -1e30f;
    for (int j = tid; j < VV; j += 256) m = fmaxf(m, y[j]);
    red[tid] = m; __syncthreads();
    for (int s = 128; s; s >>= 1) {
        if (tid < s) red[tid] = fmaxf(red[tid], red[tid + s]);
        __syncthreads();
    }
    float mx = red[0];
    __syncthreads();

    float sum = 0.f;
    for (int j = tid; j < VV; j += 256) sum += expf(y[j] - mx);
    red[tid] = sum; __syncthreads();
    for (int s = 128; s; s >>= 1) {
        if (tid < s) red[tid] += red[tid + s];
        __syncthreads();
    }
    if (tid == 0) {
        float logZ   = mx + logf(red[0]);
        int   target = (t < TO) ? tok[n * TO + t] : 2 /*EOS*/;
        nll[b] = logZ - y[target];
    }
}

__global__ __launch_bounds__(256)
void loss_reduce_kernel(const float* __restrict__ nll, float* __restrict__ out)
{
    __shared__ float red[256];
    int tid = threadIdx.x;
    float s = 0.f;
    for (int i = tid; i < NB * TT; i += 256) s += nll[i];
    red[tid] = s; __syncthreads();
    for (int k = 128; k; k >>= 1) {
        if (tid < k) red[tid] += red[tid + k];
        __syncthreads();
    }
    if (tid == 0) out[0] = red[0] / (float)(NB * TT);
}

// ---------------------------------------------------------------------------
extern "C" void kernel_launch(void* const* d_in, const int* in_sizes, int n_in,
                              void* d_out, int out_size, void* d_ws, size_t ws_size,
                              hipStream_t stream)
{
    const int*   padded_input = (const int*)  d_in[0];
    const float* enc          = (const float*)d_in[1];
    const float* emb          = (const float*)d_in[2];
    const float* W_ih0        = (const float*)d_in[3];   // [2048,1024]
    const float* W_hh0        = (const float*)d_in[4];   // [2048,512]
    const float* b_ih0        = (const float*)d_in[5];
    const float* b_hh0        = (const float*)d_in[6];
    const float* W_ih_r       = (const float*)d_in[7];   // [2,2048,512]
    const float* W_hh_r       = (const float*)d_in[8];   // [2,2048,512]
    const float* b_ih_r       = (const float*)d_in[9];   // [2,2048]
    const float* b_hh_r       = (const float*)d_in[10];
    const float* W1           = (const float*)d_in[11];  // [512,1024]
    const float* b1           = (const float*)d_in[12];
    const float* W2           = (const float*)d_in[13];  // [10000,512]
    const float* b2           = (const float*)d_in[14];

    float* ws     = (float*)d_ws;
    float* h      = ws;                 // [3][32][512] = 49152
    float* c      = ws + 49152;         // 49152
    float* attctx = ws + 98304;         // 16384
    float* rnn_in = ws + 114688;        // 32768
    float* gates  = ws + 147456;        // 65536
    float* mlp_in = ws + 212992;        // 32768
    float* hidden = ws + 245760;        // 16384
    float* nll    = ws + 262144;        // 2080

    float* y_all = (float*)d_out;       // [32][65][10000], then loss scalar

    // zero recurrent state + attention context
    zero_kernel<<<(114688 + 255) / 256, 256, 0, stream>>>(ws, 114688);

    for (int t = 0; t < TT; ++t) {
        build_rnn_in_kernel<<<128, 256, 0, stream>>>(padded_input, emb, attctx,
                                                     rnn_in, t);
        // layer 0: gates = rnn_in.Wi^T + h0.Wh^T + b   (64 waves -> 8 blocks)
        gemm_wmma_kernel<<<8, 256, 0, stream>>>(
            rnn_in, 1024, 1024, h, 512, 512,
            W_ih0, 1024, W_hh0, 512, b_ih0, b_hh0,
            gates, 2048L, 2048, 0);
        lstm_pw_kernel<<<64, 256, 0, stream>>>(gates, h, c);

        for (int l = 1; l < 3; ++l) {
            gemm_wmma_kernel<<<8, 256, 0, stream>>>(
                h + (size_t)(l - 1) * NB * HH, 512, 512,
                h + (size_t)l * NB * HH,       512, 512,
                W_ih_r + (size_t)(l - 1) * 2048 * 512, 512,
                W_hh_r + (size_t)(l - 1) * 2048 * 512, 512,
                b_ih_r + (size_t)(l - 1) * 2048,
                b_hh_r + (size_t)(l - 1) * 2048,
                gates, 2048L, 2048, 0);
            lstm_pw_kernel<<<64, 256, 0, stream>>>(gates,
                                                   h + (size_t)l * NB * HH,
                                                   c + (size_t)l * NB * HH);
        }

        attn_kernel<<<NB, 512, 0, stream>>>(enc, h + 2 * NB * HH, attctx, mlp_in);

        // hidden = tanh(mlp_in.W1^T + b1)   (16 waves -> 2 blocks)
        gemm_wmma_kernel<<<2, 256, 0, stream>>>(
            mlp_in, 1024, 1024, nullptr, 0, 0,
            W1, 1024, nullptr, 0, b1, nullptr,
            hidden, 512L, 512, 1);

        // y[:, t, :] = hidden.W2^T + b2     (313 waves -> 40 blocks)
        gemm_wmma_kernel<<<40, 256, 0, stream>>>(
            hidden, 512, 512, nullptr, 0, 0,
            W2, 512, nullptr, 0, b2, nullptr,
            y_all + (size_t)t * VV, (long)TT * VV, VV, 0);
    }

    nll_kernel<<<NB * TT, 256, 0, stream>>>(y_all, padded_input, nll);
    loss_reduce_kernel<<<1, 256, 0, stream>>>(nll,
                                              y_all + (size_t)NB * TT * VV);
}
